// MoE_51616916963811
// MI455X (gfx1250) — matrix operand order repaired and checked by reading, not compiled, not run on hardware
//
#include <hip/hip_runtime.h>
#include <hip/hip_bf16.h>

typedef unsigned short u16;
typedef unsigned int   u32;

// ---- model dims (fixed by reference) ----
#define T_TOK   2048
#define DIM     2048
#define INTER   1408
#define NEXP    16
#define TOPK    4
#define NGRP    4
#define SINTER  2816

// K chunk per staging phase (two 32-wide WMMA slices)
#define KC    64
// LDS row stride: 72 bf16 = 144B (16B-aligned rows; 36 words -> 16 distinct banks
// across any 16 consecutive rows; makes each half-fragment contiguous -> ds_load_b128)
#define KPAD2 72

typedef __attribute__((ext_vector_type(16))) __bf16 v16bf;
typedef __attribute__((ext_vector_type(8)))  float  v8f;
typedef __attribute__((ext_vector_type(4)))  float  f32x4;
typedef __attribute__((ext_vector_type(2)))  u32    u32x2;
typedef __attribute__((ext_vector_type(4)))  u32    u32x4;

union Frag { v16bf v; u32 u[8]; };

__device__ __forceinline__ u16 f32_to_bf16(float f) {
    union { float f; u32 u; } c; c.f = f;
    u32 r = c.u + 0x7FFFu + ((c.u >> 16) & 1u);   // round-to-nearest-even
    return (u16)(r >> 16);
}

#if defined(__has_builtin)
#if __has_builtin(__builtin_amdgcn_cvt_pk_bf16_f32)
#define HAVE_CVT_PK_BF16 1
#endif
#endif

__device__ __forceinline__ u32 pack2_bf16(float a, float b) {
#ifdef HAVE_CVT_PK_BF16
    auto p = __builtin_amdgcn_cvt_pk_bf16_f32(a, b);   // v_cvt_pk_bf16_f32
    return __builtin_bit_cast(u32, p);
#else
    return (u32)f32_to_bf16(a) | ((u32)f32_to_bf16(b) << 16);
#endif
}

__device__ __forceinline__ u32x2 cvt4_bf16(f32x4 v) {
    u32x2 r;
    r.x = pack2_bf16(v.x, v.y);
    r.y = pack2_bf16(v.z, v.w);
    return r;
}

__device__ __forceinline__ float fast_silu(float g) {
    // g * sigmoid(g); v_rcp_f32 instead of IEEE-div chain (bf16 output anyway)
    return g * __builtin_amdgcn_rcpf(1.0f + __expf(-g));
}

// A fragment: 16x32 bf16 slice s of [32][KC] LDS tile; rows = lane&15 (+16*mw)
__device__ __forceinline__ v16bf load_a_frag(const u16* As, int mw, int lane, int s) {
    int m    = (lane & 15) + mw * 16;
    int half = lane >> 4;
    const u32* row = (const u32*)(As + m * KPAD2 + s * 32);
    Frag f;
#pragma unroll
    for (int v = 0; v < 8; ++v) {
        int ui = (v < 4 ? v : v + 4) + 4 * half;  // K pairs: 0..7 / 16..23 (+8 for half)
        f.u[v] = row[ui];
    }
    return f.v;
}

// B fragment: 32x16 bf16 slice s of [64][KC] LDS tile; col = lane&15 (+16*nw)
__device__ __forceinline__ v16bf load_b_frag(const u16* Bs, int nw, int lane, int s) {
    int n    = (lane & 15) + nw * 16;
    int half = lane >> 4;
    const u32* row = (const u32*)(Bs + n * KPAD2 + s * 32);
    Frag f;
#pragma unroll
    for (int v = 0; v < 8; ++v)
        f.u[v] = row[(v < 4 ? v : v + 4) + 4 * half];
    return f.v;
}

// ---------------- kernel 1: zero output (vectorized) + counters ----------------
__global__ void zero_kernel(float* __restrict__ out, int* __restrict__ counts) {
    int i = blockIdx.x * blockDim.x + threadIdx.x;
    if (i < (T_TOK * DIM) / 4) {
        f32x4 z = {};
        *(f32x4*)(&out[i * 4]) = z;
    }
    if (i < NEXP) counts[i] = 0;
}

// ---------------- kernel 2: gate + routing scatter ----------------
__global__ void gate_kernel(const float* __restrict__ x, const float* __restrict__ Wg,
                            int* __restrict__ counts, int* __restrict__ tok_list,
                            float* __restrict__ wt_list) {
    __shared__ float xs[DIM];
    __shared__ float logits[NEXP];
    int t   = blockIdx.x;
    int tid = threadIdx.x;
    int lane = tid & 31, wave = tid >> 5;
#pragma unroll
    for (int i = tid; i < DIM / 4; i += 256)
        *(f32x4*)(&xs[i * 4]) = *(const f32x4*)(&x[(long)t * DIM + i * 4]);
    __syncthreads();

    // wave w handles experts 2w, 2w+1
    int e0 = 2 * wave, e1 = 2 * wave + 1;
    float p0 = 0.f, p1 = 0.f;
    for (int d = lane; d < DIM; d += 32) {
        float xv = xs[d];
        p0 += xv * Wg[e0 * DIM + d];
        p1 += xv * Wg[e1 * DIM + d];
    }
#pragma unroll
    for (int m = 16; m >= 1; m >>= 1) {
        p0 += __shfl_xor(p0, m, 32);
        p1 += __shfl_xor(p1, m, 32);
    }
    if (lane == 0) { logits[e0] = p0; logits[e1] = p1; }
    __syncthreads();

    if (tid == 0) {
        float sc[NEXP];
        float mx = -1e30f;
#pragma unroll
        for (int e = 0; e < NEXP; ++e) mx = fmaxf(mx, logits[e]);
        float sum = 0.f;
#pragma unroll
        for (int e = 0; e < NEXP; ++e) { sc[e] = __expf(logits[e] - mx); sum += sc[e]; }
        float inv = 1.f / sum;
#pragma unroll
        for (int e = 0; e < NEXP; ++e) sc[e] *= inv;

        // group scores = max over 4 experts per group
        float gs[NGRP];
#pragma unroll
        for (int g = 0; g < NGRP; ++g) {
            float m4 = sc[4 * g];
#pragma unroll
            for (int j = 1; j < 4; ++j) m4 = fmaxf(m4, sc[4 * g + j]);
            gs[g] = m4;
        }
        // top-2 groups (first occurrence on ties, like jax top_k)
        int g0 = 0;
        for (int g = 1; g < NGRP; ++g) if (gs[g] > gs[g0]) g0 = g;
        int g1 = (g0 == 0) ? 1 : 0;
        for (int g = 0; g < NGRP; ++g) if (g != g0 && gs[g] > gs[g1]) g1 = g;

        float masked[NEXP];
#pragma unroll
        for (int e = 0; e < NEXP; ++e) {
            int g = e >> 2;
            masked[e] = (g == g0 || g == g1) ? sc[e] : -1e30f;
        }
        // top-4 experts
#pragma unroll
        for (int k = 0; k < TOPK; ++k) {
            int best = 0;
            for (int e = 1; e < NEXP; ++e) if (masked[e] > masked[best]) best = e;
            masked[best] = -2e30f;
            float w = sc[best];           // ROUTE_SCALE = 1
            int pos = atomicAdd(&counts[best], 1);
            tok_list[best * T_TOK + pos] = t;
            wt_list[best * T_TOK + pos]  = w;
        }
    }
}

// ---------------- kernel 3: exclusive scan over 16 counters ----------------
__global__ void scan_kernel(const int* __restrict__ counts, int* __restrict__ offsets) {
    if (threadIdx.x == 0) {
        int acc = 0;
        offsets[0] = 0;
#pragma unroll
        for (int e = 0; e < NEXP; ++e) { acc += counts[e]; offsets[e + 1] = acc; }
    }
}

// ---------------- kernel 4: routed up/gate proj + SwiGLU ----------------
// grid: (INTER/64, T/32, NEXP), block 256 (8 waves -> 2x4 grid of 16x16 WMMA tiles)
__global__ void routed_up_kernel(const float* __restrict__ x,
                                 const float* __restrict__ W1,
                                 const float* __restrict__ W3,
                                 const int* __restrict__ tok_list,
                                 const int* __restrict__ counts,
                                 const int* __restrict__ offsets,
                                 u16* __restrict__ H) {
    int e   = blockIdx.z;
    int cnt = counts[e];
    int row0 = blockIdx.y * 32;
    if (row0 >= cnt) return;
    int n0 = blockIdx.x * 64;

    __shared__ u16 As[32 * KPAD2];
    __shared__ u16 B1s[64 * KPAD2];
    __shared__ u16 B3s[64 * KPAD2];
    __shared__ int toks[32];

    int tid = threadIdx.x, lane = tid & 31, wave = tid >> 5;
    int mw = wave >> 2, nw = wave & 3;

    if (tid < 32) {
        int r = row0 + tid;
        toks[tid] = tok_list[e * T_TOK + (r < cnt ? r : 0)];
    }
    __syncthreads();

    long wbase = (long)e * INTER * DIM;
    v8f acc1 = {}; v8f acc3 = {};
    for (int k0 = 0; k0 < DIM; k0 += KC) {
        {   // A: 32 rows x 16 float4 -> b128 loads, b64 LDS stores
            int i = tid;
#pragma unroll
            for (int it = 0; it < 2; ++it, i += 256) {
                int m = i >> 4, q = i & 15;
                f32x4 v = *(const f32x4*)(&x[(long)toks[m] * DIM + k0 + q * 4]);
                *(u32x2*)(&As[m * KPAD2 + q * 4]) = cvt4_bf16(v);
            }
        }
        {   // B1,B3: 64 rows x 16 float4 each
            int i = tid;
#pragma unroll
            for (int it = 0; it < 4; ++it, i += 256) {
                int n = i >> 4, q = i & 15;
                long widx = wbase + (long)(n0 + n) * DIM + k0 + q * 4;
                *(u32x2*)(&B1s[n * KPAD2 + q * 4]) = cvt4_bf16(*(const f32x4*)(&W1[widx]));
                *(u32x2*)(&B3s[n * KPAD2 + q * 4]) = cvt4_bf16(*(const f32x4*)(&W3[widx]));
                __builtin_prefetch(&W1[widx + KC], 0, 1);
                __builtin_prefetch(&W3[widx + KC], 0, 1);
            }
        }
        __syncthreads();
#pragma unroll
        for (int s = 0; s < 2; ++s) {
            v16bf a  = load_a_frag(As,  mw, lane, s);
            v16bf b1 = load_b_frag(B1s, nw, lane, s);
            v16bf b3 = load_b_frag(B3s, nw, lane, s);
            acc1 = __builtin_amdgcn_wmma_f32_16x16x32_bf16(false, a, false, b1, (short)0, acc1, false, false);
            acc3 = __builtin_amdgcn_wmma_f32_16x16x32_bf16(false, a, false, b3, (short)0, acc3, false, false);
        }
        __syncthreads();
    }
    int base = offsets[e];
    int half = lane >> 4;
    int n = (lane & 15) + nw * 16 + n0;
#pragma unroll
    for (int r = 0; r < 8; ++r) {
        int m  = r + 8 * half + mw * 16;
        int rg = row0 + m;
        if (rg < cnt) {
            float hv = fast_silu(acc1[r]) * acc3[r];
            H[(long)(base + rg) * INTER + n] = f32_to_bf16(hv);
        }
    }
}

// ---------------- kernel 5: routed down proj + weighted combine ----------------
// grid: (DIM/64, T/32, NEXP)
__global__ void routed_down_kernel(const u16* __restrict__ H,
                                   const float* __restrict__ W2,
                                   const int* __restrict__ tok_list,
                                   const float* __restrict__ wt_list,
                                   const int* __restrict__ counts,
                                   const int* __restrict__ offsets,
                                   float* __restrict__ out) {
    int e   = blockIdx.z;
    int cnt = counts[e];
    int row0 = blockIdx.y * 32;
    if (row0 >= cnt) return;
    int d0 = blockIdx.x * 64;

    __shared__ u16 As[32 * KPAD2];
    __shared__ u16 Bs[64 * KPAD2];

    int tid = threadIdx.x, lane = tid & 31, wave = tid >> 5;
    int mw = wave >> 2, nw = wave & 3;
    int base = offsets[e];
    long wbase = (long)e * DIM * INTER;

    v8f acc = {};
    for (int k0 = 0; k0 < INTER; k0 += KC) {
        {   // A: H rows (already bf16), 32 rows x 8 x (8 bf16) -> b128 copies
            int m = tid >> 3, q = tid & 7;
            int rg = row0 + m; if (rg >= cnt) rg = cnt - 1;
            *(u32x4*)(&As[m * KPAD2 + q * 8]) =
                *(const u32x4*)(&H[(long)(base + rg) * INTER + k0 + q * 8]);
        }
        {   // B: W2[e] 64 rows x 16 float4
            int i = tid;
#pragma unroll
            for (int it = 0; it < 4; ++it, i += 256) {
                int n = i >> 4, q = i & 15;
                long widx = wbase + (long)(d0 + n) * INTER + k0 + q * 4;
                *(u32x2*)(&Bs[n * KPAD2 + q * 4]) = cvt4_bf16(*(const f32x4*)(&W2[widx]));
                __builtin_prefetch(&W2[widx + KC], 0, 1);
            }
        }
        __syncthreads();
#pragma unroll
        for (int s = 0; s < 2; ++s) {
            v16bf a = load_a_frag(As, mw, lane, s);
            v16bf b = load_b_frag(Bs, nw, lane, s);
            acc = __builtin_amdgcn_wmma_f32_16x16x32_bf16(false, a, false, b, (short)0, acc, false, false);
        }
        __syncthreads();
    }
    int half = lane >> 4;
    int d = (lane & 15) + nw * 16 + d0;
#pragma unroll
    for (int r = 0; r < 8; ++r) {
        int m  = r + 8 * half + mw * 16;
        int rg = row0 + m;
        if (rg < cnt) {
            int   tok = tok_list[e * T_TOK + rg];
            float w   = wt_list[e * T_TOK + rg];
            atomicAdd(&out[(long)tok * DIM + d], w * acc[r]);
        }
    }
}

// ---------------- kernel 6: shared expert up + SwiGLU ----------------
// grid: (SINTER/64, T/32)
__global__ void shared_up_kernel(const float* __restrict__ x,
                                 const float* __restrict__ Ws1,
                                 const float* __restrict__ Ws3,
                                 u16* __restrict__ Hs) {
    int row0 = blockIdx.y * 32;
    int n0   = blockIdx.x * 64;
    __shared__ u16 As[32 * KPAD2];
    __shared__ u16 B1s[64 * KPAD2];
    __shared__ u16 B3s[64 * KPAD2];

    int tid = threadIdx.x, lane = tid & 31, wave = tid >> 5;
    int mw = wave >> 2, nw = wave & 3;

    v8f acc1 = {}; v8f acc3 = {};
    for (int k0 = 0; k0 < DIM; k0 += KC) {
        {
            int i = tid;
#pragma unroll
            for (int it = 0; it < 2; ++it, i += 256) {
                int m = i >> 4, q = i & 15;
                f32x4 v = *(const f32x4*)(&x[(long)(row0 + m) * DIM + k0 + q * 4]);
                *(u32x2*)(&As[m * KPAD2 + q * 4]) = cvt4_bf16(v);
            }
        }
        {
            int i = tid;
#pragma unroll
            for (int it = 0; it < 4; ++it, i += 256) {
                int n = i >> 4, q = i & 15;
                long widx = (long)(n0 + n) * DIM + k0 + q * 4;
                *(u32x2*)(&B1s[n * KPAD2 + q * 4]) = cvt4_bf16(*(const f32x4*)(&Ws1[widx]));
                *(u32x2*)(&B3s[n * KPAD2 + q * 4]) = cvt4_bf16(*(const f32x4*)(&Ws3[widx]));
                __builtin_prefetch(&Ws1[widx + KC], 0, 1);
                __builtin_prefetch(&Ws3[widx + KC], 0, 1);
            }
        }
        __syncthreads();
#pragma unroll
        for (int s = 0; s < 2; ++s) {
            v16bf a  = load_a_frag(As,  mw, lane, s);
            v16bf b1 = load_b_frag(B1s, nw, lane, s);
            v16bf b3 = load_b_frag(B3s, nw, lane, s);
            acc1 = __builtin_amdgcn_wmma_f32_16x16x32_bf16(false, a, false, b1, (short)0, acc1, false, false);
            acc3 = __builtin_amdgcn_wmma_f32_16x16x32_bf16(false, a, false, b3, (short)0, acc3, false, false);
        }
        __syncthreads();
    }
    int half = lane >> 4;
    int n = (lane & 15) + nw * 16 + n0;
#pragma unroll
    for (int r = 0; r < 8; ++r) {
        int m = r + 8 * half + mw * 16;
        float hv = fast_silu(acc1[r]) * acc3[r];
        Hs[(long)(row0 + m) * SINTER + n] = f32_to_bf16(hv);
    }
}

// ---------------- kernel 7: shared expert down, plain accumulate ----------------
// grid: (DIM/64, T/32); runs last, each (t,d) touched by exactly one thread
__global__ void shared_down_kernel(const u16* __restrict__ Hs,
                                   const float* __restrict__ Ws2,
                                   float* __restrict__ out) {
    int row0 = blockIdx.y * 32;
    int d0   = blockIdx.x * 64;
    __shared__ u16 As[32 * KPAD2];
    __shared__ u16 Bs[64 * KPAD2];

    int tid = threadIdx.x, lane = tid & 31, wave = tid >> 5;
    int mw = wave >> 2, nw = wave & 3;

    v8f acc = {};
    for (int k0 = 0; k0 < SINTER; k0 += KC) {
        {   // A: Hs rows (bf16) -> b128 copies
            int m = tid >> 3, q = tid & 7;
            *(u32x4*)(&As[m * KPAD2 + q * 8]) =
                *(const u32x4*)(&Hs[(long)(row0 + m) * SINTER + k0 + q * 8]);
        }
        {
            int i = tid;
#pragma unroll
            for (int it = 0; it < 4; ++it, i += 256) {
                int n = i >> 4, q = i & 15;
                long widx = (long)(d0 + n) * SINTER + k0 + q * 4;
                *(u32x2*)(&Bs[n * KPAD2 + q * 4]) = cvt4_bf16(*(const f32x4*)(&Ws2[widx]));
                __builtin_prefetch(&Ws2[widx + KC], 0, 1);
            }
        }
        __syncthreads();
#pragma unroll
        for (int s = 0; s < 2; ++s) {
            v16bf a = load_a_frag(As, mw, lane, s);
            v16bf b = load_b_frag(Bs, nw, lane, s);
            acc = __builtin_amdgcn_wmma_f32_16x16x32_bf16(false, a, false, b, (short)0, acc, false, false);
        }
        __syncthreads();
    }
    int half = lane >> 4;
    int d = (lane & 15) + nw * 16 + d0;
#pragma unroll
    for (int r = 0; r < 8; ++r) {
        int m = r + 8 * half + mw * 16;
        int t = row0 + m;
        out[(long)t * DIM + d] += acc[r];   // after routed atomics (stream-ordered)
    }
}

// ---------------- host launcher ----------------
extern "C" void kernel_launch(void* const* d_in, const int* in_sizes, int n_in,
                              void* d_out, int out_size, void* d_ws, size_t ws_size,
                              hipStream_t stream) {
    const float* x   = (const float*)d_in[0];
    const float* Wg  = (const float*)d_in[1];
    const float* W1  = (const float*)d_in[2];
    const float* W3  = (const float*)d_in[3];
    const float* W2  = (const float*)d_in[4];
    const float* Ws1 = (const float*)d_in[5];
    const float* Ws3 = (const float*)d_in[6];
    const float* Ws2 = (const float*)d_in[7];
    float* out = (float*)d_out;

    char* ws = (char*)d_ws;
    int*   counts  = (int*)(ws + 0);                       // 16 ints
    int*   offsets = (int*)(ws + 64);                      // 17 ints
    int*   toks    = (int*)(ws + 256);                     // NEXP*T ints
    float* wts     = (float*)(ws + 256 + (size_t)NEXP * T_TOK * 4);
    u16*   Hr      = (u16*)(ws + 262400);                  // 8192 x 1408 bf16 (~23 MB)
    u16*   Hs      = (u16*)(ws + 23331072);                // 2048 x 2816 bf16 (~11.5 MB)

    zero_kernel<<<dim3((T_TOK * DIM / 4 + 255) / 256), dim3(256), 0, stream>>>(out, counts);
    gate_kernel<<<dim3(T_TOK), dim3(256), 0, stream>>>(x, Wg, counts, toks, wts);
    scan_kernel<<<dim3(1), dim3(32), 0, stream>>>(counts, offsets);
    routed_up_kernel<<<dim3(INTER / 64, T_TOK / 32, NEXP), dim3(256), 0, stream>>>(
        x, W1, W3, toks, counts, offsets, Hr);
    routed_down_kernel<<<dim3(DIM / 64, T_TOK / 32, NEXP), dim3(256), 0, stream>>>(
        Hr, W2, toks, wts, counts, offsets, out);
    shared_up_kernel<<<dim3(SINTER / 64, T_TOK / 32), dim3(256), 0, stream>>>(x, Ws1, Ws3, Hs);
    shared_down_kernel<<<dim3(DIM / 64, T_TOK / 32), dim3(256), 0, stream>>>(Hs, Ws2, out);
}
